// DASH_DA_TV_57836029608352
// MI455X (gfx1250) — matrix operand order repaired and checked by reading, not compiled, not run on hardware
//
#include <hip/hip_runtime.h>
#include <hip/hip_bf16.h>
#include <math.h>

typedef __attribute__((ext_vector_type(16))) _Float16 v16h;
typedef __attribute__((ext_vector_type(8)))  _Float16 v8h;
typedef __attribute__((ext_vector_type(8)))  float    v8f;

// ---------------- constants ----------------
#define BATCH 32
#define HH 32
#define WWID 32
#define CDIM 384
#define NHEAD 12
#define HDIM 32
#define LTOK 1024           // H*W
#define WS7 7
#define NTOK 49             // WS*WS
#define NWIN 800            // B * 5 * 5
#define MQ 39200            // NWIN * NTOK  (mult of 16)
#define MX 32768            // B * L       (mult of 16)

// ---------------- helpers ----------------
__device__ __forceinline__ v16h ld16h(const _Float16* p0, const _Float16* p1) {
  v8h lo = *(const v8h*)p0;
  v8h hi = *(const v8h*)p1;
  v16h r;
#pragma unroll
  for (int i = 0; i < 8; ++i) { r[i] = lo[i]; r[8 + i] = hi[i]; }
  return r;
}

__device__ __forceinline__ v16h zero16h() {
  v16h r;
#pragma unroll
  for (int i = 0; i < 16; ++i) r[i] = (_Float16)0.f;
  return r;
}

// Wave-level 16x64 tile GEMM core. All 10 fragment loads of a k-tile are
// issued BEFORE the 4 WMMAs so the compiler emits one load clause + graduated
// s_wait_loadcnt (loads complete in order), and the next k-tile's clause
// overlaps the XDL pipe. No manual double-buffer => no spills (~80 live VGPRs).
// A: row-major f16 (MxK).  Bp packed fragments:
//   Bp[((kt*NT + nt)*32 + lane)*16 + e] = B[kt*32 + 16*(lane/16)+e][nt*16 + lane%16]
template<int K, int NT>
__device__ __forceinline__ void gemm_wave(const _Float16* __restrict__ A,
                                          const _Float16* __restrict__ Bp,
                                          int mt, int ng, int lane, v8f acc[4]) {
  const int half = lane >> 4, l16 = lane & 15;
  const _Float16* ar = A + (size_t)(mt * 16 + l16) * K + half * 16;
  const _Float16* br = Bp + (((size_t)ng * 4) * 32 + lane) * 16;
  for (int kt = 0; kt < K / 32; ++kt) {
    v16h a  = ld16h(ar, ar + 8);
    v16h b0 = ld16h(br,        br + 8);
    v16h b1 = ld16h(br + 512,  br + 520);
    v16h b2 = ld16h(br + 1024, br + 1032);
    v16h b3 = ld16h(br + 1536, br + 1544);
    acc[0] = __builtin_amdgcn_wmma_f32_16x16x32_f16(false, a, false, b0, (short)0, acc[0], false, false);
    acc[1] = __builtin_amdgcn_wmma_f32_16x16x32_f16(false, a, false, b1, (short)0, acc[1], false, false);
    acc[2] = __builtin_amdgcn_wmma_f32_16x16x32_f16(false, a, false, b2, (short)0, acc[2], false, false);
    acc[3] = __builtin_amdgcn_wmma_f32_16x16x32_f16(false, a, false, b3, (short)0, acc[3], false, false);
    ar += 32;
    br += (size_t)NT * 32 * 16;
  }
}

// ---------------- weight repack: f32 (KxN row-major) -> packed f16 fragments ----------------
__global__ void pack_b_kernel(const float* __restrict__ B, _Float16* __restrict__ out,
                              int K, int N) {
  int idx = blockIdx.x * 256 + threadIdx.x;
  if (idx >= K * N) return;
  int e    = idx & 15;
  int lane = (idx >> 4) & 31;
  int tile = idx >> 9;
  int NT = N >> 4;
  int nt = tile % NT, kt = tile / NT;
  int k = kt * 32 + (lane >> 4) * 16 + e;
  int n = nt * 16 + (lane & 15);
  out[idx] = (_Float16)B[(size_t)k * N + n];
}

// ---------------- LN1 + window partition (pad windows -> LN(0) rows) ----------------
__global__ void ln1_window_kernel(const float* __restrict__ x, const float* __restrict__ w,
                                  const float* __restrict__ b, _Float16* __restrict__ out) {
  int row  = blockIdx.x * 8 + (threadIdx.x >> 5);   // 0..MQ-1
  int lane = threadIdx.x & 31;
  int win = row / NTOK, tok = row % NTOK;
  int bb = win / 25, wr = win % 25;
  int wh = wr / 5, wwi = wr % 5;
  int h = wh * WS7 + tok / WS7, wc = wwi * WS7 + tok % WS7;
  bool valid = (h < HH) && (wc < WWID);
  const float* src = x + ((size_t)bb * LTOK + h * WWID + wc) * CDIM;
  float v[12];
  float s = 0.f;
#pragma unroll
  for (int i = 0; i < 12; ++i) { v[i] = valid ? src[lane + 32 * i] : 0.f; s += v[i]; }
#pragma unroll
  for (int m = 1; m < 32; m <<= 1) s += __shfl_xor(s, m, 32);
  float mu = s * (1.f / CDIM);
  float vs = 0.f;
#pragma unroll
  for (int i = 0; i < 12; ++i) { float d = v[i] - mu; vs += d * d; }
#pragma unroll
  for (int m = 1; m < 32; m <<= 1) vs += __shfl_xor(vs, m, 32);
  float rs = rsqrtf(vs * (1.f / CDIM) + 1e-5f);
#pragma unroll
  for (int i = 0; i < 12; ++i) {
    int c = lane + 32 * i;
    out[(size_t)row * CDIM + c] = (_Float16)((v[i] - mu) * rs * w[c] + b[c]);
  }
}

// ---------------- LN2 (plain rows) ----------------
__global__ void ln2_kernel(const float* __restrict__ x, const float* __restrict__ w,
                           const float* __restrict__ b, _Float16* __restrict__ out) {
  int row  = blockIdx.x * 8 + (threadIdx.x >> 5);   // 0..MX-1
  int lane = threadIdx.x & 31;
  const float* src = x + (size_t)row * CDIM;
  float v[12];
  float s = 0.f;
#pragma unroll
  for (int i = 0; i < 12; ++i) { v[i] = src[lane + 32 * i]; s += v[i]; }
#pragma unroll
  for (int m = 1; m < 32; m <<= 1) s += __shfl_xor(s, m, 32);
  float mu = s * (1.f / CDIM);
  float vs = 0.f;
#pragma unroll
  for (int i = 0; i < 12; ++i) { float d = v[i] - mu; vs += d * d; }
#pragma unroll
  for (int m = 1; m < 32; m <<= 1) vs += __shfl_xor(vs, m, 32);
  float rs = rsqrtf(vs * (1.f / CDIM) + 1e-5f);
#pragma unroll
  for (int i = 0; i < 12; ++i) {
    int c = lane + 32 * i;
    out[(size_t)row * CDIM + c] = (_Float16)((v[i] - mu) * rs * w[c] + b[c]);
  }
}

// ---------------- bias expand: (NH, 49) gathered by (49,49) idx -> (NH,49,49) f32 ----------------
__global__ void bias_expand_kernel(const float* __restrict__ ab, const int* __restrict__ idx,
                                   float* __restrict__ be) {
  int i = blockIdx.x * 256 + threadIdx.x;
  if (i >= NHEAD * NTOK * NTOK) return;
  int h = i / (NTOK * NTOK), nm = i % (NTOK * NTOK);
  be[i] = ab[h * NTOK + idx[nm]];
}

// ---------------- GEMM kernels ----------------
__global__ void gemm_qkv_kernel(const _Float16* __restrict__ A, const _Float16* __restrict__ Bp,
                                const float* __restrict__ bias, _Float16* __restrict__ out) {
  const int N = 3 * CDIM, NT = N / 16, NG = N / 64, MT = MQ / 16;
  int wave = blockIdx.x * 8 + (threadIdx.x >> 5);
  int mt = wave % MT, ng = wave / MT;
  if (ng >= NG) return;
  int lane = threadIdx.x & 31, half = lane >> 4, l16 = lane & 15;
  v8f acc[4];
#pragma unroll
  for (int t = 0; t < 4; ++t)
#pragma unroll
    for (int r = 0; r < 8; ++r) acc[t][r] = 0.f;
  gemm_wave<CDIM, NT>(A, Bp, mt, ng, lane, acc);
#pragma unroll
  for (int t = 0; t < 4; ++t) {
    int col = (ng * 4 + t) * 16 + l16;
    float bz = bias[col];
#pragma unroll
    for (int r = 0; r < 8; ++r) {
      int row = mt * 16 + r + 8 * half;
      out[(size_t)row * N + col] = (_Float16)(acc[t][r] + bz);
    }
  }
}

__global__ void gemm_proj_kernel(const _Float16* __restrict__ A, const _Float16* __restrict__ Bp,
                                 const float* __restrict__ bias, const float* __restrict__ xin,
                                 float* __restrict__ x1) {
  const int N = CDIM, NT = N / 16, NG = N / 64, MT = MQ / 16;
  int wave = blockIdx.x * 8 + (threadIdx.x >> 5);
  int mt = wave % MT, ng = wave / MT;
  if (ng >= NG) return;
  int lane = threadIdx.x & 31, half = lane >> 4, l16 = lane & 15;
  v8f acc[4];
#pragma unroll
  for (int t = 0; t < 4; ++t)
#pragma unroll
    for (int r = 0; r < 8; ++r) acc[t][r] = 0.f;
  gemm_wave<CDIM, NT>(A, Bp, mt, ng, lane, acc);
#pragma unroll
  for (int t = 0; t < 4; ++t) {
    int col = (ng * 4 + t) * 16 + l16;
    float pb = bias[col];
#pragma unroll
    for (int r = 0; r < 8; ++r) {
      int row = mt * 16 + r + 8 * half;
      int win = row / NTOK, tok = row % NTOK;
      int bb = win / 25, wr = win % 25;
      int wh = wr / 5, wwi = wr % 5;
      int h = wh * WS7 + tok / WS7, wc = wwi * WS7 + tok % WS7;
      if (h < HH && wc < WWID) {
        size_t o = ((size_t)bb * LTOK + h * WWID + wc) * CDIM + col;
        x1[o] = acc[t][r] + pb + xin[o];
      }
    }
  }
}

__global__ void gemm_fc1_kernel(const _Float16* __restrict__ A, const _Float16* __restrict__ Bp,
                                const float* __restrict__ bias, _Float16* __restrict__ out) {
  const int N = 4 * CDIM, NT = N / 16, NG = N / 64, MT = MX / 16;
  int wave = blockIdx.x * 8 + (threadIdx.x >> 5);
  int mt = wave % MT, ng = wave / MT;
  if (ng >= NG) return;
  int lane = threadIdx.x & 31, half = lane >> 4, l16 = lane & 15;
  v8f acc[4];
#pragma unroll
  for (int t = 0; t < 4; ++t)
#pragma unroll
    for (int r = 0; r < 8; ++r) acc[t][r] = 0.f;
  gemm_wave<CDIM, NT>(A, Bp, mt, ng, lane, acc);
#pragma unroll
  for (int t = 0; t < 4; ++t) {
    int col = (ng * 4 + t) * 16 + l16;
    float bz = bias[col];
#pragma unroll
    for (int r = 0; r < 8; ++r) {
      int row = mt * 16 + r + 8 * half;
      float v = acc[t][r] + bz;
      float g = 0.5f * v * (1.f + erff(v * 0.70710678118654752f));  // exact GELU
      out[(size_t)row * N + col] = (_Float16)g;
    }
  }
}

__global__ void gemm_fc2_kernel(const _Float16* __restrict__ A, const _Float16* __restrict__ Bp,
                                const float* __restrict__ bias, const float* __restrict__ res,
                                float* __restrict__ out) {
  const int N = CDIM, NT = N / 16, NG = N / 64, MT = MX / 16, K = 4 * CDIM;
  int wave = blockIdx.x * 8 + (threadIdx.x >> 5);
  int mt = wave % MT, ng = wave / MT;
  if (ng >= NG) return;
  int lane = threadIdx.x & 31, half = lane >> 4, l16 = lane & 15;
  v8f acc[4];
#pragma unroll
  for (int t = 0; t < 4; ++t)
#pragma unroll
    for (int r = 0; r < 8; ++r) acc[t][r] = 0.f;
  gemm_wave<K, NT>(A, Bp, mt, ng, lane, acc);
#pragma unroll
  for (int t = 0; t < 4; ++t) {
    int col = (ng * 4 + t) * 16 + l16;
    float bz = bias[col];
#pragma unroll
    for (int r = 0; r < 8; ++r) {
      int row = mt * 16 + r + 8 * half;
      size_t o = (size_t)row * N + col;
      out[o] = acc[t][r] + bz + res[o];
    }
  }
}

// ---------------- windowed attention: one block per (window, head), 4 waves ----------------
__global__ void attn_kernel(const _Float16* __restrict__ qkv, const float* __restrict__ biasE,
                            _Float16* __restrict__ oin) {
  __shared__ __align__(16) _Float16 Psh[4][16][80];   // per-wave 16x64 probs, padded
  int blk = blockIdx.x;
  int win = blk / NHEAD, hd = blk % NHEAD;
  int wv = threadIdx.x >> 5;                 // 0..3 -> M tile
  int lane = threadIdx.x & 31;
  int half = lane >> 4, l16 = lane & 15;
  int m0 = wv * 16;
  const int QKVN = 3 * CDIM;

  // ---- logits = Q Kt  (A = Q rows, B = Kt) ----
  v16h aq;
  {
    int tq = m0 + l16;
    if (tq < NTOK) {
      const _Float16* p = qkv + (size_t)(win * NTOK + tq) * QKVN + hd * 96 + half * 16;
      aq = ld16h(p, p + 8);
    } else aq = zero16h();
  }
  v8f acc[4];
#pragma unroll
  for (int t = 0; t < 4; ++t)
#pragma unroll
    for (int r = 0; r < 8; ++r) acc[t][r] = 0.f;
#pragma unroll
  for (int t = 0; t < 4; ++t) {
    v16h bk;
    int tk = t * 16 + l16;
    if (tk < NTOK) {
      const _Float16* p = qkv + (size_t)(win * NTOK + tk) * QKVN + hd * 96 + 32 + half * 16;
      bk = ld16h(p, p + 8);
    } else bk = zero16h();
    acc[t] = __builtin_amdgcn_wmma_f32_16x16x32_f16(false, aq, false, bk,
                                                    (short)0, acc[t], false, false);
  }

  // ---- prefetch V fragments early: their gather latency hides behind softmax ----
  v16h bv[2][2];
#pragma unroll
  for (int ks = 0; ks < 2; ++ks)
#pragma unroll
    for (int nt = 0; nt < 2; ++nt) {
#pragma unroll
      for (int e = 0; e < 16; ++e) {
        int tok = ks * 32 + 16 * half + e;
        int d = nt * 16 + l16;
        bv[ks][nt][e] = (tok < NTOK)
                          ? qkv[(size_t)(win * NTOK + tok) * QKVN + hd * 96 + 64 + d]
                          : (_Float16)0.f;
      }
    }

  // ---- softmax over 49 keys (rows live in 16-lane halves; 4 cols/lane) ----
  const float scale = 0.17677669529663687f;  // 1/sqrt(32)
#pragma unroll
  for (int r = 0; r < 8; ++r) {
    int m = m0 + r + 8 * half;
    float vv[4];
    float mx = -1e30f;
#pragma unroll
    for (int t = 0; t < 4; ++t) {
      int col = t * 16 + l16;
      float z = -1e30f;
      if (m < NTOK && col < NTOK)
        z = acc[t][r] * scale + biasE[((size_t)hd * NTOK + m) * NTOK + col];
      vv[t] = z;
      mx = fmaxf(mx, z);
    }
#pragma unroll
    for (int msk = 1; msk < 16; msk <<= 1) mx = fmaxf(mx, __shfl_xor(mx, msk, 32));
    float se = 0.f;
#pragma unroll
    for (int t = 0; t < 4; ++t) {
      float e = (vv[t] > -1e29f) ? __expf(vv[t] - mx) : 0.f;
      vv[t] = e;
      se += e;
    }
#pragma unroll
    for (int msk = 1; msk < 16; msk <<= 1) se += __shfl_xor(se, msk, 32);
    float inv = (se > 0.f) ? 1.f / se : 0.f;
#pragma unroll
    for (int t = 0; t < 4; ++t)
      Psh[wv][r + 8 * half][t * 16 + l16] = (_Float16)(vv[t] * inv);
  }

  // ---- O = P V  (A = P from LDS, B = prefetched V) ----
  v8f acc2[2];
#pragma unroll
  for (int n = 0; n < 2; ++n)
#pragma unroll
    for (int r = 0; r < 8; ++r) acc2[n][r] = 0.f;
#pragma unroll
  for (int ks = 0; ks < 2; ++ks) {
    const _Float16* pr = &Psh[wv][l16][ks * 32 + half * 16];
    v16h ap = ld16h(pr, pr + 8);
#pragma unroll
    for (int nt = 0; nt < 2; ++nt)
      acc2[nt] = __builtin_amdgcn_wmma_f32_16x16x32_f16(false, ap, false, bv[ks][nt],
                                                        (short)0, acc2[nt], false, false);
  }
#pragma unroll
  for (int nt = 0; nt < 2; ++nt)
#pragma unroll
    for (int r = 0; r < 8; ++r) {
      int m = m0 + r + 8 * half;
      if (m < NTOK)
        oin[(size_t)(win * NTOK + m) * CDIM + hd * HDIM + nt * 16 + l16] =
            (_Float16)acc2[nt][r];
    }
}

// ---------------- depthwise 3x3 conv + BN (one pixel per block, 384 threads) ----------------
__global__ void conv_bn_kernel(const float* __restrict__ x1, const float* __restrict__ cw,
                               const float* __restrict__ g, const float* __restrict__ bb,
                               const float* __restrict__ mn, const float* __restrict__ vr,
                               float* __restrict__ x2) {
  int pix = blockIdx.x;
  int c = threadIdx.x;
  int b = pix >> 10;
  int hw = pix & 1023;
  int h = hw >> 5, w = hw & 31;
  float s = 0.f;
#pragma unroll
  for (int dh = -1; dh <= 1; ++dh)
#pragma unroll
    for (int dw = -1; dw <= 1; ++dw) {
      int hh = h + dh, ww = w + dw;
      if (hh >= 0 && hh < HH && ww >= 0 && ww < WWID)
        s += x1[((size_t)b * LTOK + hh * WWID + ww) * CDIM + c] *
             cw[c * 9 + (dh + 1) * 3 + (dw + 1)];
    }
  float inv = rsqrtf(vr[c] + 1e-5f);
  x2[((size_t)b * LTOK + h * WWID + w) * CDIM + c] = (s - mn[c]) * inv * g[c] + bb[c];
}

// ---------------- host launch ----------------
extern "C" void kernel_launch(void* const* d_in, const int* in_sizes, int n_in,
                              void* d_out, int out_size, void* d_ws, size_t ws_size,
                              hipStream_t stream) {
  (void)in_sizes; (void)n_in; (void)out_size; (void)ws_size;
  const float* x       = (const float*)d_in[0];
  const float* ln1_w   = (const float*)d_in[1];
  const float* ln1_b   = (const float*)d_in[2];
  const float* qkv_w   = (const float*)d_in[3];
  const float* qkv_b   = (const float*)d_in[4];
  const float* proj_w  = (const float*)d_in[5];
  const float* proj_b  = (const float*)d_in[6];
  const float* attn_b  = (const float*)d_in[7];
  const int*   bidx    = (const int*)d_in[8];
  const float* conv_w  = (const float*)d_in[9];
  const float* bn_g    = (const float*)d_in[10];
  const float* bn_b    = (const float*)d_in[11];
  const float* bn_m    = (const float*)d_in[12];
  const float* bn_v    = (const float*)d_in[13];
  const float* ln2_w   = (const float*)d_in[14];
  const float* ln2_b   = (const float*)d_in[15];
  const float* fc1_w   = (const float*)d_in[16];
  const float* fc1_b   = (const float*)d_in[17];
  const float* fc2_w   = (const float*)d_in[18];
  const float* fc2_b   = (const float*)d_in[19];
  float* out = (float*)d_out;

  char* ws = (char*)d_ws;
  size_t off = 0;
  auto alloc = [&](size_t bytes) -> void* {
    void* p = ws + off;
    off = (off + bytes + 255) & ~(size_t)255;
    return p;
  };
  _Float16* wq   = (_Float16*)alloc((size_t)CDIM * 3 * CDIM * 2);
  _Float16* wp   = (_Float16*)alloc((size_t)CDIM * CDIM * 2);
  _Float16* w1   = (_Float16*)alloc((size_t)CDIM * 4 * CDIM * 2);
  _Float16* w2   = (_Float16*)alloc((size_t)4 * CDIM * CDIM * 2);
  _Float16* xw   = (_Float16*)alloc((size_t)MQ * CDIM * 2);
  _Float16* qkv  = (_Float16*)alloc((size_t)MQ * 3 * CDIM * 2);
  float*    bE   = (float*)   alloc((size_t)NHEAD * NTOK * NTOK * 4);
  _Float16* oin  = (_Float16*)alloc((size_t)MQ * CDIM * 2);
  float*    x1   = (float*)   alloc((size_t)MX * CDIM * 4);
  float*    x2   = (float*)   alloc((size_t)MX * CDIM * 4);
  _Float16* h16  = (_Float16*)alloc((size_t)MX * CDIM * 2);
  _Float16* hd16 = (_Float16*)alloc((size_t)MX * 4 * CDIM * 2);

  // weight repack (weights are tiny; done every call, deterministic)
  pack_b_kernel<<<(CDIM * 3 * CDIM + 255) / 256, 256, 0, stream>>>(qkv_w, wq, CDIM, 3 * CDIM);
  pack_b_kernel<<<(CDIM * CDIM + 255) / 256, 256, 0, stream>>>(proj_w, wp, CDIM, CDIM);
  pack_b_kernel<<<(CDIM * 4 * CDIM + 255) / 256, 256, 0, stream>>>(fc1_w, w1, CDIM, 4 * CDIM);
  pack_b_kernel<<<(4 * CDIM * CDIM + 255) / 256, 256, 0, stream>>>(fc2_w, w2, 4 * CDIM, CDIM);
  bias_expand_kernel<<<(NHEAD * NTOK * NTOK + 255) / 256, 256, 0, stream>>>(attn_b, bidx, bE);

  // LN1 + window partition
  ln1_window_kernel<<<MQ / 8, 256, 0, stream>>>(x, ln1_w, ln1_b, xw);

  // QKV GEMM: (39200x384)x(384x1152)
  gemm_qkv_kernel<<<((MQ / 16) * (3 * CDIM / 64) + 7) / 8, 256, 0, stream>>>(xw, wq, qkv_b, qkv);

  // windowed attention, one block per (window, head)
  attn_kernel<<<NWIN * NHEAD, 128, 0, stream>>>(qkv, bE, oin);

  // proj GEMM + un-window + crop + residual
  gemm_proj_kernel<<<((MQ / 16) * (CDIM / 64) + 7) / 8, 256, 0, stream>>>(oin, wp, proj_b, x, x1);

  // depthwise conv + BN
  conv_bn_kernel<<<BATCH * LTOK, CDIM, 0, stream>>>(x1, conv_w, bn_g, bn_b, bn_m, bn_v, x2);

  // LN2
  ln2_kernel<<<MX / 8, 256, 0, stream>>>(x2, ln2_w, ln2_b, h16);

  // fc1 GEMM + exact GELU
  gemm_fc1_kernel<<<((MX / 16) * (4 * CDIM / 64) + 7) / 8, 256, 0, stream>>>(h16, w1, fc1_b, hd16);

  // fc2 GEMM + residual -> out
  gemm_fc2_kernel<<<((MX / 16) * (CDIM / 64) + 7) / 8, 256, 0, stream>>>(hd16, w2, fc2_b, x2, out);
}
